// PSAR_19301583028775
// MI455X (gfx1250) — matrix operand order repaired and checked by reading, not compiled, not run on hardware
//
#include <hip/hip_runtime.h>

typedef _Float16 h16;
typedef __attribute__((ext_vector_type(16))) _Float16 v16h;
typedef __attribute__((ext_vector_type(8)))  _Float16 v8h;
typedef __attribute__((ext_vector_type(4)))  _Float16 v4h;
typedef __attribute__((ext_vector_type(8)))  float    v8f;

// ---------------------------------------------------------------------------
// f32 -> f16 weight convert with K-padding: dst[row*Kp + k] = (k<K)?src[row*K+k]:0
// Padded rows make WMMA A-operand loads guard-free and 16B-aligned.
// ---------------------------------------------------------------------------
__global__ void k_cvt_pad(const float* __restrict__ src, h16* __restrict__ dst,
                          int K, int Kp, int total) {
    int i = blockIdx.x * blockDim.x + threadIdx.x;
    if (i >= total) return;
    int row = i / Kp, k = i - row * Kp;
    dst[i] = (k < K) ? (h16)src[row * K + k] : (h16)0.f;
}

// upc bias + MEAN fold
__global__ void k_bias3(const float* __restrict__ ub, float* __restrict__ o) {
    int c = threadIdx.x;
    if (c < 3) {
        const float M[3] = {0.4488f, 0.4371f, 0.4040f};
        o[c] = ub[c] + M[c];
    }
}

// ---------------------------------------------------------------------------
// sub_mean + patch decomposition (-> p [100,3,32,32] f16) + image copy into
// mbuf channels 0..2 ([4,67,160,160] f16)
// ---------------------------------------------------------------------------
__global__ void k_prep(const float* __restrict__ x, h16* __restrict__ p,
                       h16* __restrict__ mbuf) {
    int i = blockIdx.x * blockDim.x + threadIdx.x;
    const int TOT = 4 * 3 * 160 * 160;
    if (i >= TOT) return;
    int xw = i % 160; int t = i / 160;
    int y  = t % 160; t /= 160;
    int c  = t % 3;   int b = t / 3;
    const float M[3] = {0.4488f, 0.4371f, 0.4040f};
    h16 h = (h16)(x[i] - M[c]);
    mbuf[(((size_t)b * 67 + c) * 160 + y) * 160 + xw] = h;
    int n = b * 25 + (y >> 5) * 5 + (xw >> 5);
    p[(((size_t)n * 3 + c) * 32 + (y & 31)) * 32 + (xw & 31)] = h;
}

// ---------------------------------------------------------------------------
// A-operand fetch: weights padded to Kp per row -> unconditional vector loads.
// Per-lane layout: half h -> K = h + 16*hi, i.e. two contiguous
// 8-half (16B) runs at k0+16*hi and k0+16*hi+8.
// ---------------------------------------------------------------------------
__device__ __forceinline__
v16h load_A(const h16* __restrict__ wgt, int arow, int Cout, int Kp,
            int k0, int hi) {
    int arowc = (arow < Cout) ? arow : (Cout - 1);   // clamp; junk rows unstored
    const v8h* wrow = (const v8h*)(wgt + (size_t)arowc * Kp + k0);
    v8h a0 = wrow[2 * hi];
    v8h a1 = wrow[2 * hi + 1];
    v16h av;
#pragma unroll
    for (int h = 0; h < 8; ++h) { av[h] = a0[h]; av[h + 8] = a1[h]; }
    return av;
}

// ---------------------------------------------------------------------------
// Shared epilogue: bias (+opt residual f16) (+opt relu), store
// f16 plain / f16 pixel-shuffle-2 remap / f32 plain.
// C/D f32 layout: vgpr r -> M = r + 8*(lane>>4), N = lane&15.
// ---------------------------------------------------------------------------
__device__ __forceinline__
void conv_store(v8f acc, int co0, int hi, int pix, int py, int px,
                const float* __restrict__ bias,
                const h16* __restrict__ res, int resC,
                h16* __restrict__ outH, float* __restrict__ outF,
                int outC, int coOff, int img, int H, int W, int Cout,
                int relu, int ps)
{
    const int HWp = H * W;
#pragma unroll
    for (int r = 0; r < 8; ++r) {
        int co = co0 + r + (hi << 3);
        if (co >= Cout || pix >= HWp) continue;
        float v = acc[r] + bias[co];
        if (res) v += (float)res[(((size_t)img * resC + co) * H + py) * W + px];
        if (relu) v = fmaxf(v, 0.f);
        if (outF) {
            outF[(((size_t)img * outC + coOff + co) * H + py) * W + px] = v;
        } else if (ps) {
            int c2 = co >> 2, ry = (co >> 1) & 1, rx = co & 1;
            outH[(((size_t)img * outC + coOff + c2) * (2 * H) + (2 * py + ry)) *
                     (2 * W) + (2 * px + rx)] = (h16)v;
        } else {
            outH[(((size_t)img * outC + coOff + co) * H + py) * W + px] = (h16)v;
        }
    }
}

// ---------------------------------------------------------------------------
// 1-wave implicit-GEMM conv via V_WMMA_F32_16X16X32_F16 (grouped / small Cout).
// B: lane holds N=lane&15, half h -> K = h + 16*(lane>>4).
// Branch-free gathers: 32-bit in-image offset + cndmask select; loads batched.
// ---------------------------------------------------------------------------
template <int KW, int PAD>
__global__ void __launch_bounds__(32)
k_conv1(const h16* __restrict__ in, int inC, int Cin,
        const h16* __restrict__ wgt, int Kp, const float* __restrict__ bias,
        const h16* __restrict__ res, int resC,
        h16* __restrict__ outH, float* __restrict__ outF,
        int outC, int coOff, int H, int W, int Cout, int groups,
        int relu, int ps)
{
    constexpr int KK = KW * KW;
    const int lane = threadIdx.x & 31;
    const int mrow = lane & 15;
    const int hi   = lane >> 4;
    const int HWp  = H * W;
    const int pix  = blockIdx.x * 16 + mrow;
    const int co0  = blockIdx.y * 16;
    const int img  = blockIdx.z;
    const int CinG = Cin / groups;
    const int Ktot = CinG * KK;
    const int grp  = (groups == 4) ? blockIdx.y : 0;  // Cout/groups==16 tiles
    const int cinB = grp * CinG;
    const int py   = pix / W;
    const int px   = pix - py * W;
    const int pval = (pix < HWp);
    const int arow = co0 + mrow;
    const h16* __restrict__ ibase = in + ((size_t)img * inC + cinB) * HWp;

    v8f acc = {};
    for (int k0 = 0; k0 < Ktot; k0 += 32) {
        v16h av = load_A(wgt, arow, Cout, Kp, k0, hi);

        h16 tv[16];
        int ok[16];
#pragma unroll
        for (int h = 0; h < 16; ++h) {
            int kb = k0 + h + (hi << 4);
            int kv = (kb < Ktot);
            int kbc = kv ? kb : 0;
            int ci = kbc / KK; int rr = kbc - ci * KK;
            int ky = rr / KW;  int kx = rr - ky * KW;
            int yy = py + ky - PAD, xx = px + kx - PAD;
            int o = kv & pval & ((unsigned)yy < (unsigned)H) &
                    ((unsigned)xx < (unsigned)W);
            int idx = (ci * H + yy) * W + xx;     // fits in 32 bits
            idx = o ? idx : 0;
            tv[h] = ibase[idx];
            ok[h] = o;
        }
        v16h bv;
#pragma unroll
        for (int h = 0; h < 16; ++h) bv[h] = ok[h] ? tv[h] : (h16)0.f;

        acc = __builtin_amdgcn_wmma_f32_16x16x32_f16(
                  false, av, false, bv, (short)0, acc, false, false);
    }
    conv_store(acc, co0, hi, pix, py, px, bias, res, resC, outH, outF,
               outC, coOff, img, H, W, Cout, relu, ps);
}

// ---------------------------------------------------------------------------
// 4-wave implicit-GEMM conv, groups==1, Cout multiple-of-64 tiles.
// 32x16 B tile gathered once per K-chunk into LDS ([N][K], K contiguous),
// shared by 4 co-tiles: each thread fills 4 consecutive-K halves with one
// ds_store_b64; each lane consumes two aligned ds_load_b128. Weights
// prefetched a chunk ahead (global_prefetch_b8). Requires W % 16 == 0.
// ---------------------------------------------------------------------------
template <int KW, int PAD>
__global__ void __launch_bounds__(128)
k_conv4(const h16* __restrict__ in, int inC, int Cin,
        const h16* __restrict__ wgt, int Kp, const float* __restrict__ bias,
        h16* __restrict__ outH, float* __restrict__ outF,
        int outC, int coOff, int H, int W, int Cout,
        int relu, int ps)
{
    constexpr int KK = KW * KW;
    __shared__ __align__(32) h16 Bs[16 * 32];          // [N][K]
    const int tid  = threadIdx.x;
    const int wave = tid >> 5;
    const int lane = tid & 31;
    const int mrow = lane & 15;
    const int hi   = lane >> 4;
    const int pix0 = blockIdx.x * 16;
    const int co0  = blockIdx.y * 64 + wave * 16;
    const int img  = blockIdx.z;
    const int Ktot = Cin * KK;
    const int py0  = pix0 / W;                 // tile within one row (W%16==0)
    const int px0  = pix0 - py0 * W;
    const int arow = co0 + mrow;
    const int fn   = tid >> 3;                 // fill: N index 0..15
    const int fk   = (tid & 7) << 2;           // fill: K base (4 consecutive)
    const h16* __restrict__ ibase = in + (size_t)img * inC * H * W;

    v8f acc = {};
    for (int k0 = 0; k0 < Ktot; k0 += 32) {
        // --- cooperative B gather: 4 consecutive-K halves per thread ---
        h16 tv[4];
        int ok[4];
#pragma unroll
        for (int e = 0; e < 4; ++e) {
            int kb = k0 + fk + e;
            int kv = (kb < Ktot);
            int kbc = kv ? kb : 0;
            int ci = kbc / KK; int rr = kbc - ci * KK;
            int ky = rr / KW;  int kx = rr - ky * KW;
            int yy = py0 + ky - PAD, xx = px0 + fn + kx - PAD;
            int o = kv & ((unsigned)yy < (unsigned)H) &
                    ((unsigned)xx < (unsigned)W);
            int idx = (ci * H + yy) * W + xx;   // fits in 32 bits
            idx = o ? idx : 0;
            tv[e] = ibase[idx];
            ok[e] = o;
        }
        v4h pk;
#pragma unroll
        for (int e = 0; e < 4; ++e) pk[e] = ok[e] ? tv[e] : (h16)0.f;
        *(v4h*)(Bs + fn * 32 + fk) = pk;
        __syncthreads();

        // prefetch next chunk of this row's weights into GL2
        __builtin_prefetch(wgt + (size_t)arow * Kp + k0 + 32, 0, 3);

        v16h av = load_A(wgt, arow, Cout, Kp, k0, hi);

        const v8h* bp = (const v8h*)(Bs + mrow * 32 + (hi << 4));
        v8h b0 = bp[0], b1 = bp[1];
        v16h bv;
#pragma unroll
        for (int h = 0; h < 8; ++h) { bv[h] = b0[h]; bv[h + 8] = b1[h]; }

        acc = __builtin_amdgcn_wmma_f32_16x16x32_f16(
                  false, av, false, bv, (short)0, acc, false, false);
        __syncthreads();
    }
    conv_store(acc, co0, hi, pix0 + mrow, py0, px0 + mrow, bias,
               nullptr, 0, outH, outF, outC, coOff, img, H, W, Cout, relu, ps);
}

// ---------------------------------------------------------------------------
// ssza[n] = mean |f1[n]| over 12*1024
// ---------------------------------------------------------------------------
__global__ void k_ssza(const h16* __restrict__ f1, float* __restrict__ ssza) {
    int n = blockIdx.x;
    const h16* base = f1 + (size_t)n * 12 * 1024;
    float s = 0.f;
    for (int i = threadIdx.x; i < 12 * 1024; i += blockDim.x)
        s += fabsf((float)base[i]);
    __shared__ float sm[256];
    sm[threadIdx.x] = s; __syncthreads();
    for (int st = 128; st > 0; st >>= 1) {
        if (threadIdx.x < st) sm[threadIdx.x] += sm[threadIdx.x + st];
        __syncthreads();
    }
    if (threadIdx.x == 0) ssza[n] = sm[0] / (12.f * 1024.f);
}

// std of channel-0 patch from original f32 input (shift-invariant)
__global__ void k_std(const float* __restrict__ x, float* __restrict__ sstd) {
    int n = blockIdx.x;
    int b = n / 25, r = n % 25, ph = r / 5, pw = r % 5;
    const float* base = x + (size_t)b * 3 * 160 * 160 + (size_t)ph * 32 * 160 + pw * 32;
    float s = 0.f, s2 = 0.f;
    for (int i = threadIdx.x; i < 1024; i += blockDim.x) {
        float v = base[(i >> 5) * 160 + (i & 31)];
        s += v; s2 += v * v;
    }
    __shared__ float sa[256], sb[256];
    sa[threadIdx.x] = s; sb[threadIdx.x] = s2; __syncthreads();
    for (int st = 128; st > 0; st >>= 1) {
        if (threadIdx.x < st) { sa[threadIdx.x] += sa[threadIdx.x + st];
                                sb[threadIdx.x] += sb[threadIdx.x + st]; }
        __syncthreads();
    }
    if (threadIdx.x == 0) {
        float m = sa[0] / 1024.f;
        sstd[n] = sqrtf(fmaxf(sb[0] / 1024.f - m * m, 0.f));
    }
}

// threshold routing + patch stitch into mbuf channels 3..66
__global__ void k_route_stitch(const h16* __restrict__ xs, const h16* __restrict__ xm,
                               const h16* __restrict__ xh, const float* __restrict__ ssza,
                               const float* __restrict__ tl, const float* __restrict__ tr,
                               h16* __restrict__ mbuf) {
    int i = blockIdx.x * blockDim.x + threadIdx.x;
    const int TOT = 4 * 64 * 160 * 160;
    if (i >= TOT) return;
    int xw = i % 160; int t = i / 160;
    int y  = t % 160; t /= 160;
    int c  = t % 64;  int b = t / 64;
    int n = b * 25 + (y >> 5) * 5 + (xw >> 5);
    float sa = ssza[n];
    float tl0 = tl[0], thr = tl0 + tr[0];
    const h16* src = (sa > thr) ? xs : ((sa >= tl0) ? xm : xh);
    h16 v = src[(((size_t)n * 64 + c) * 32 + (y & 31)) * 32 + (xw & 31)];
    mbuf[(((size_t)b * 67 + 3 + c) * 160 + y) * 160 + xw] = v;
}

__global__ void k_timep(const float* __restrict__ ssza, const float* __restrict__ tl,
                        const float* __restrict__ tr, float* __restrict__ tp) {
    int b = blockIdx.x, t = threadIdx.x;
    float tl0 = tl[0], tr0 = tr[0], thr = tl0 + tr0;
    float v = 0.f;
    if (t < 25) {
        float sa = ssza[b * 25 + t];
        if (sa > thr)        v = 1e-4f * (sa - thr);
        else if (sa >= tl0)  v = 2e-4f * tr0;
        else                 v = 3e-4f * (tl0 - sa);
    }
    __shared__ float sm[32];
    sm[t] = v; __syncthreads();
    for (int st = 16; st > 0; st >>= 1) {
        if (t < st) sm[t] += sm[t + st];
        __syncthreads();
    }
    if (t == 0) tp[b] = sm[0] / 25.f;
}

__global__ void k_xcropia(const float* __restrict__ sstd, float* __restrict__ xc) {
    int i = blockIdx.x * blockDim.x + threadIdx.x;
    const int TOT = 4 * 640 * 640;
    if (i >= TOT) return;
    int X = i % 640; int t = i / 640;
    int Y = t % 640; int b = t / 640;
    xc[i] = sstd[b * 25 + (Y / 128) * 5 + (X / 128)];
}

__global__ void k_xoutputa(const float* __restrict__ y, const float* __restrict__ xc,
                           float* __restrict__ xa) {
    int i = blockIdx.x * blockDim.x + threadIdx.x;
    const int TOT = 4 * 3 * 640 * 640;
    if (i >= TOT) return;
    int X = i % 640; int t = i / 640;
    int Y = t % 640; t /= 640;
    int b = t / 3;
    xa[i] = y[i] * xc[(size_t)b * 409600 + (size_t)Y * 640 + X];
}

// ---------------------------------------------------------------------------
extern "C" void kernel_launch(void* const* d_in, const int* in_sizes, int n_in,
                              void* d_out, int out_size, void* d_ws, size_t ws_size,
                              hipStream_t stream)
{
    (void)in_sizes; (void)n_in; (void)out_size; (void)ws_size;
    char* ws = (char*)d_ws;
    auto al = [](size_t x) { return (x + 255) & ~(size_t)255; };

    // region A (patch phase), overlapped later by y3
    size_t aoff = 0;
    auto AA = [&](size_t bytes) { size_t o = aoff; aoff += al(bytes); return o; };
    const size_t o_p    = AA((size_t)100 * 3 * 1024 * 2);
    const size_t o_f1   = AA((size_t)100 * 12 * 1024 * 2);
    const size_t o_cbuf = AA((size_t)100 * 256 * 1024 * 2);
    const size_t o_tmp  = AA((size_t)100 * 64 * 1024 * 2);
    const size_t o_xs   = AA((size_t)100 * 64 * 1024 * 2);
    const size_t o_xm   = AA((size_t)100 * 64 * 1024 * 2);
    const size_t o_xh   = AA((size_t)100 * 64 * 1024 * 2);
    const size_t y3_bytes = (size_t)4 * 64 * 409600 * 2;
    size_t regA = (aoff > y3_bytes) ? aoff : y3_bytes;

    size_t off = al(regA);
    auto AB = [&](size_t bytes) { size_t o = off; off += al(bytes); return o; };
    const size_t o_mbuf = AB((size_t)4 * 67 * 25600 * 2);
    const size_t o_y1   = AB((size_t)4 * 64 * 25600 * 2);
    const size_t o_y2   = AB((size_t)4 * 64 * 102400 * 2);
    const size_t o_ssza = AB(100 * 4);
    const size_t o_sstd = AB(100 * 4);
    const size_t o_b3   = AB(3 * 4);

    h16*   p     = (h16*)(ws + o_p);
    h16*   f1    = (h16*)(ws + o_f1);
    h16*   cbuf  = (h16*)(ws + o_cbuf);
    h16*   tmp   = (h16*)(ws + o_tmp);
    h16*   xs    = (h16*)(ws + o_xs);
    h16*   xm    = (h16*)(ws + o_xm);
    h16*   xh    = (h16*)(ws + o_xh);
    h16*   y3    = (h16*)(ws);            // overlaps region A (phase A dead)
    h16*   mbuf  = (h16*)(ws + o_mbuf);
    h16*   y1    = (h16*)(ws + o_y1);
    h16*   y2    = (h16*)(ws + o_y2);
    float* ssza  = (float*)(ws + o_ssza);
    float* sstd  = (float*)(ws + o_sstd);
    float* bias3 = (float*)(ws + o_b3);

    // f16 weight arena, rows padded to Kp (multiple of 32 halves)
    const int widx[15]  = {3, 5, 7, 9, 11, 13, 15, 17, 19, 21, 23, 25, 27, 29, 31};
    const int wrows[15] = {12, 64, 64, 64, 64, 64, 64, 64, 64, 64, 64, 64, 256, 256, 3};
    const int wK[15]    = {27, 108, 144, 144, 128, 144, 144, 192, 144, 144, 256, 603, 576, 576, 576};
    const int wKp[15]   = {32, 128, 160, 160, 128, 160, 160, 192, 160, 160, 256, 608, 576, 576, 576};
    h16* wp[15];
    for (int i = 0; i < 15; ++i) {
        int total = wrows[i] * wKp[i];
        size_t o = AB((size_t)total * 2);
        wp[i] = (h16*)(ws + o);
        k_cvt_pad<<<(total + 255) / 256, 256, 0, stream>>>(
            (const float*)d_in[widx[i]], wp[i], wK[i], wKp[i], total);
    }
    const float* bI[15];
    const int bidx[15] = {4, 6, 8, 10, 12, 14, 16, 18, 20, 22, 24, 26, 28, 30, 32};
    for (int i = 0; i < 15; ++i) bI[i] = (const float*)d_in[bidx[i]];

    float* out_y  = (float*)d_out;
    float* out_tp = out_y + (size_t)4915200;
    float* out_xc = out_tp + 4;
    float* out_xa = out_xc + (size_t)1638400;

    // 1-wave conv (grouped / small Cout), 3x3 pad 1
    auto conv1_3 = [&](const h16* in, int inC, int Cin, const h16* w, int Kp,
                       const float* b, const h16* res, int resC, h16* oH, float* oF,
                       int outC, int coOff, int N, int H, int W, int Cout, int g,
                       int relu, int ps) {
        dim3 grid((H * W + 15) / 16, (Cout + 15) / 16, N);
        k_conv1<3, 1><<<grid, 32, 0, stream>>>(in, inC, Cin, w, Kp, b, res, resC,
                                               oH, oF, outC, coOff, H, W, Cout, g,
                                               relu, ps);
    };
    // 4-wave LDS-shared conv, groups=1
    auto conv4_3 = [&](const h16* in, int inC, int Cin, const h16* w, int Kp,
                       const float* b, h16* oH, float* oF, int outC, int coOff,
                       int N, int H, int W, int Cout, int relu, int ps) {
        dim3 grid((H * W) / 16, Cout / 64, N);
        k_conv4<3, 1><<<grid, 128, 0, stream>>>(in, inC, Cin, w, Kp, b, oH, oF,
                                                outC, coOff, H, W, Cout, relu, ps);
    };
    auto conv4_1 = [&](const h16* in, int inC, int Cin, const h16* w, int Kp,
                       const float* b, h16* oH, float* oF, int outC, int coOff,
                       int N, int H, int W, int Cout, int relu, int ps) {
        dim3 grid((H * W) / 16, Cout / 64, N);
        k_conv4<1, 0><<<grid, 128, 0, stream>>>(in, inC, Cin, w, Kp, b, oH, oF,
                                                outC, coOff, H, W, Cout, relu, ps);
    };

    k_bias3<<<1, 32, 0, stream>>>((const float*)d_in[32], bias3);
    k_prep<<<(307200 + 255) / 256, 256, 0, stream>>>((const float*)d_in[0], p, mbuf);

    // IICM + score
    conv1_3(p, 3, 3, wp[0], 32, bI[0], nullptr, 0, f1, nullptr, 12, 0,
            100, 32, 32, 12, 1, 1, 0);
    k_ssza<<<100, 256, 0, stream>>>(f1, ssza);
    conv4_3(f1, 12, 12, wp[1], 128, bI[1], cbuf, nullptr, 256, 0,
            100, 32, 32, 64, 0, 0);
    // block 1 (grouped 3x3, residual in 2nd conv)
    conv1_3(cbuf, 256, 64, wp[2], 160, bI[2], nullptr, 0, tmp, nullptr, 64, 0,
            100, 32, 32, 64, 4, 1, 0);
    conv1_3(tmp, 64, 64, wp[3], 160, bI[3], cbuf, 256, cbuf, nullptr, 256, 64,
            100, 32, 32, 64, 4, 0, 0);
    conv4_1(cbuf, 256, 128, wp[4], 128, bI[4], xs, nullptr, 64, 0,
            100, 32, 32, 64, 1, 0);
    // block 2
    conv1_3(xs, 64, 64, wp[5], 160, bI[5], nullptr, 0, tmp, nullptr, 64, 0,
            100, 32, 32, 64, 4, 1, 0);
    conv1_3(tmp, 64, 64, wp[6], 160, bI[6], xs, 64, cbuf, nullptr, 256, 128,
            100, 32, 32, 64, 4, 0, 0);
    conv4_1(cbuf, 256, 192, wp[7], 192, bI[7], xm, nullptr, 64, 0,
            100, 32, 32, 64, 1, 0);
    // block 3
    conv1_3(xm, 64, 64, wp[8], 160, bI[8], nullptr, 0, tmp, nullptr, 64, 0,
            100, 32, 32, 64, 4, 1, 0);
    conv1_3(tmp, 64, 64, wp[9], 160, bI[9], xm, 64, cbuf, nullptr, 256, 192,
            100, 32, 32, 64, 4, 0, 0);
    conv4_1(cbuf, 256, 256, wp[10], 256, bI[10], xh, nullptr, 64, 0,
            100, 32, 32, 64, 1, 0);

    // routing / stats
    k_std<<<100, 256, 0, stream>>>((const float*)d_in[0], sstd);
    k_timep<<<4, 32, 0, stream>>>(ssza, (const float*)d_in[1],
                                  (const float*)d_in[2], out_tp);
    k_route_stitch<<<(6553600 + 255) / 256, 256, 0, stream>>>(
        xs, xm, xh, ssza, (const float*)d_in[1], (const float*)d_in[2], mbuf);

    // MARM + upsampling (pixel-shuffle fused in epilogue)
    conv4_3(mbuf, 67, 67, wp[11], 608, bI[11], y1, nullptr, 64, 0,
            4, 160, 160, 64, 1, 0);
    conv4_3(y1, 64, 64, wp[12], 576, bI[12], y2, nullptr, 64, 0,
            4, 160, 160, 256, 1, 1);
    conv4_3(y2, 64, 64, wp[13], 576, bI[13], y3, nullptr, 64, 0,
            4, 320, 320, 256, 1, 1);
    conv1_3(y3, 64, 64, wp[14], 576, bias3, nullptr, 0, nullptr, out_y, 3, 0,
            4, 640, 640, 3, 1, 0, 0);

    k_xcropia<<<(1638400 + 255) / 256, 256, 0, stream>>>(sstd, out_xc);
    k_xoutputa<<<(4915200 + 255) / 256, 256, 0, stream>>>(out_y, out_xc, out_xa);
}
